// QuantizedLinear_8667244003662
// MI455X (gfx1250) — matrix operand order repaired and checked by reading, not compiled, not run on hardware
//
#include <hip/hip_runtime.h>
#include <stdint.h>

// ---- problem dims (fixed by the reference) ----
#define TOKENS 4096
#define D_IN   4096
#define D_OUT  16384

// ---- tiling ----
#define BM 128
#define BN 256
#define BK 64
#define LDA (BK + 8)   // padded LDS row stride (halves); 72*2=144B, 16B-aligned
#define LDB (BK + 8)

typedef __bf16  v16bf __attribute__((ext_vector_type(16)));
typedef __bf16  v8bf  __attribute__((ext_vector_type(8)));
typedef float   v8f   __attribute__((ext_vector_type(8)));
typedef float   v4f   __attribute__((ext_vector_type(4)));
typedef int8_t  v16c  __attribute__((ext_vector_type(16)));

__global__ __launch_bounds__(256)
void qlinear_bf16_wmma(const float* __restrict__ X,
                       const int8_t* __restrict__ W,
                       const float* __restrict__ scale,
                       const float* __restrict__ bias,
                       float* __restrict__ Out)
{
    __shared__ __bf16 sA[BM * LDA];   // x tile, bf16            (18 KB)
    __shared__ __bf16 sB[BN * LDB];   // W tile, bf16 from int8  (36 KB)

    const int tid  = threadIdx.x;
    const int lane = tid & 31;
    const int wave = tid >> 5;

    const int block_n = blockIdx.x * BN;
    const int block_m = blockIdx.y * BM;

    // 8 waves: 2 along M x 4 along N; each wave owns a 64x64 macro-tile
    const int waveM = (wave & 1) * 64;
    const int waveN = (wave >> 1) * 64;

    // ---- staging assignment ----
    // A: 128 rows x 64 fp32 -> thread t: row t/2, 32 consecutive floats
    const int arow = tid >> 1;
    const int acol = (tid & 1) * 32;
    const float* srcA = X + (size_t)(block_m + arow) * D_IN + acol;
    __bf16* dstA = &sA[arow * LDA + acol];
    // B: 256 rows x 64 int8 -> thread t: row t, 64 consecutive bytes
    const int brow = tid;
    const int8_t* srcW = W + (size_t)(block_n + brow) * D_IN;
    __bf16* dstB = &sB[brow * LDB];

    v8f acc[4][4] = {};   // 4 M-subtiles x 4 N-subtiles of 16x16 f32

    for (int k0 = 0; k0 < D_IN; k0 += BK) {
        // ---- stage A: 32 fp32 -> 32 bf16 per thread ----
        #pragma unroll
        for (int v = 0; v < 4; ++v) {
            v4f f0 = *(const v4f*)(srcA + k0 + v * 8);
            v4f f1 = *(const v4f*)(srcA + k0 + v * 8 + 4);
            v8bf h;
            #pragma unroll
            for (int i = 0; i < 4; ++i) {
                h[i]     = (__bf16)f0[i];
                h[i + 4] = (__bf16)f1[i];
            }
            *(v8bf*)(dstA + v * 8) = h;
        }
        // ---- stage W: 64 int8 -> 64 bf16 per thread (exact) ----
        #pragma unroll
        for (int v = 0; v < 4; ++v) {
            v16c c = *(const v16c*)(srcW + k0 + v * 16);
            v8bf h0, h1;
            #pragma unroll
            for (int i = 0; i < 8; ++i) {
                h0[i] = (__bf16)(float)c[i];
                h1[i] = (__bf16)(float)c[i + 8];
            }
            *(v8bf*)(dstB + v * 16)     = h0;
            *(v8bf*)(dstB + v * 16 + 8) = h1;
        }
        // prefetch next K tile into cache while we compute this one
        if (k0 + BK < D_IN) {
            __builtin_prefetch(srcA + k0 + BK, 0, 1);
            __builtin_prefetch(srcW + k0 + BK, 0, 1);
        }
        __syncthreads();

        // ---- compute: 2 WMMA K-steps of 32; 16 WMMA each ----
        #pragma unroll
        for (int kk = 0; kk < BK; kk += 32) {
            v16bf afrag[4];
            v16bf bfrag[4];

            // A 16x32 bf16 layout: lanes 0-15 -> K 0-7 & 16-23; lanes 16-31 -> K 8-15 & 24-31
            const int k0a = kk + ((lane < 16) ? 0 : 8);
            #pragma unroll
            for (int mi = 0; mi < 4; ++mi) {
                const int row = waveM + mi * 16 + (lane & 15);
                v8bf lo = *(const v8bf*)(&sA[row * LDA + k0a]);
                v8bf hi = *(const v8bf*)(&sA[row * LDA + k0a + 16]);
                #pragma unroll
                for (int i = 0; i < 8; ++i) {
                    afrag[mi][i]     = lo[i];
                    afrag[mi][i + 8] = hi[i];
                }
            }
            // B 32x16 bf16 layout: lanes 0-15 -> K 0-7 & 16-23; lanes 16-31 -> K 8-15 & 24-31
            const int k0b = kk + ((lane < 16) ? 0 : 8);
            #pragma unroll
            for (int ni = 0; ni < 4; ++ni) {
                const int row = waveN + ni * 16 + (lane & 15);  // W row == output col n
                v8bf lo = *(const v8bf*)(&sB[row * LDB + k0b]);
                v8bf hi = *(const v8bf*)(&sB[row * LDB + k0b + 16]);
                #pragma unroll
                for (int i = 0; i < 8; ++i) {
                    bfrag[ni][i]     = lo[i];
                    bfrag[ni][i + 8] = hi[i];
                }
            }

            #pragma unroll
            for (int mi = 0; mi < 4; ++mi)
                #pragma unroll
                for (int ni = 0; ni < 4; ++ni)
                    acc[mi][ni] = __builtin_amdgcn_wmma_f32_16x16x32_bf16(
                        false, afrag[mi], false, bfrag[ni],
                        (short)0, acc[mi][ni], false, false);
        }
        __syncthreads();
    }

    // ---- epilogue: out = acc * scale[n] + bias[n] ----
    // C/D layout: VGPR r holds M = r + (lane>=16 ? 8 : 0), N = lane%16
    #pragma unroll
    for (int ni = 0; ni < 4; ++ni) {
        const int n = block_n + waveN + ni * 16 + (lane & 15);
        const float s = scale[n];
        const float b = bias[n];
        #pragma unroll
        for (int mi = 0; mi < 4; ++mi) {
            const int mbase = block_m + waveM + mi * 16 + ((lane >> 4) << 3);
            #pragma unroll
            for (int r = 0; r < 8; ++r) {
                Out[(size_t)(mbase + r) * D_OUT + n] = acc[mi][ni][r] * s + b;
            }
        }
    }
}

extern "C" void kernel_launch(void* const* d_in, const int* in_sizes, int n_in,
                              void* d_out, int out_size, void* d_ws, size_t ws_size,
                              hipStream_t stream) {
    const float*  x  = (const float*)d_in[0];
    const int8_t* qw = (const int8_t*)d_in[1];
    const float*  sc = (const float*)d_in[2];
    const float*  bs = (const float*)d_in[3];
    float* out = (float*)d_out;

    dim3 grid(D_OUT / BN, TOKENS / BM);  // (64, 32)
    qlinear_bf16_wmma<<<grid, 256, 0, stream>>>(x, qw, sc, bs, out);
}
